// SparseMoE_Cross_Attention_5111011083046
// MI455X (gfx1250) — matrix operand order repaired and checked by reading, not compiled, not run on hardware
//
#include <hip/hip_runtime.h>
#include <hip/hip_bf16.h>
#include <math.h>

#define B_TOK 4096
#define DIM   1024
#define NEXP  8
#define NHEAD 8
#define HDIM  128
#define D3    3072

#define SA    72            // LDS row stride (elems) for 64-wide K chunk, padded
#define TILE_M 128
#define TILE_N 128
#define KC    64
#define LDS_TILE (TILE_M * SA)

typedef __attribute__((ext_vector_type(8)))  float  v8f;
typedef __attribute__((ext_vector_type(16))) __bf16 v16bf;

union Frag { v16bf v; uint4 q[2]; };

__device__ __forceinline__ float bf2f(__hip_bfloat16 h) { return __bfloat162float(h); }
__device__ __forceinline__ __hip_bfloat16 f2bf(float f) { return __float2bfloat16(f); }

// ---- CDNA5 async global->LDS copy (16B per lane), tracked by ASYNCcnt ----
__device__ __forceinline__ void async_ld16(const __hip_bfloat16* g,
                                           const __hip_bfloat16* l) {
  uint32_t lds_off = (uint32_t)(uintptr_t)l;       // low 32 bits of generic = LDS offset
  asm volatile("global_load_async_to_lds_b128 %0, %1, off"
               :: "v"(lds_off), "v"((uint64_t)(uintptr_t)g)
               : "memory");
}
template <int N>
__device__ __forceinline__ void wait_async() {
  asm volatile("s_wait_asynccnt %0" :: "i"(N) : "memory");
}

__device__ __forceinline__ Frag ldA(const __hip_bfloat16* Asb, int base) {
  Frag f;
  f.q[0] = *(const uint4*)(Asb + base);
  f.q[1] = *(const uint4*)(Asb + base + 8);
  return f;
}
__device__ __forceinline__ Frag ldB(const __hip_bfloat16* Bsb, int base) {
  Frag f;
  f.q[0] = *(const uint4*)(Bsb + base);
  f.q[1] = *(const uint4*)(Bsb + base + 8);
  return f;
}
#define WMMA_BF16(A, B, C) \
  __builtin_amdgcn_wmma_f32_16x16x32_bf16(false, (A).v, false, (B).v, (short)0, (C), false, false)

// ---------------- elementwise f32 -> bf16 ----------------
__global__ void cvt_f32_bf16(const float* __restrict__ src,
                             __hip_bfloat16* __restrict__ dst, int n) {
  int i = (blockIdx.x * 256 + threadIdx.x) * 4;
  if (i < n) {
    float4 v = *(const float4*)(src + i);
    union { __hip_bfloat16 h[4]; uint2 u; } p;
    p.h[0] = f2bf(v.x); p.h[1] = f2bf(v.y); p.h[2] = f2bf(v.z); p.h[3] = f2bf(v.w);
    *(uint2*)(dst + i) = p.u;
  }
}

// ---------- tiled transpose + convert: src R x C f32 -> dst C x R bf16 ----------
__global__ void tcvt_f32_bf16(const float* __restrict__ src,
                              __hip_bfloat16* __restrict__ dst, int R, int C) {
  __shared__ float t[32][33];
  size_t zoff = (size_t)blockIdx.z * R * C;
  src += zoff; dst += zoff;
  int c0 = blockIdx.x * 32, r0 = blockIdx.y * 32;
  int tx = threadIdx.x, ty = threadIdx.y;
#pragma unroll
  for (int i = 0; i < 4; ++i)
    t[ty + 8 * i][tx] = src[(size_t)(r0 + ty + 8 * i) * C + c0 + tx];
  __syncthreads();
#pragma unroll
  for (int i = 0; i < 4; ++i)
    dst[(size_t)(c0 + ty + 8 * i) * R + r0 + tx] = f2bf(t[tx][ty + 8 * i]);
}

// ---------------- gating: softmax + bias + top-2, bucket counts ----------------
__global__ void gate_topk(const float* __restrict__ x, const float* __restrict__ Wg,
                          const float* __restrict__ bg, const float* __restrict__ ebias,
                          float* __restrict__ gw, int* __restrict__ ge,
                          int* __restrict__ cnt) {
  int wid = threadIdx.x >> 5, lane = threadIdx.x & 31;
  int b = blockIdx.x * 8 + wid;
  float acc[8] = {0, 0, 0, 0, 0, 0, 0, 0};
  const float* xr = x + (size_t)b * DIM;
  for (int d = lane; d < DIM; d += 32) {
    float xv = xr[d];
    const float4* w = (const float4*)(Wg + (size_t)d * NEXP);
    float4 a = w[0], c = w[1];
    acc[0] += xv * a.x; acc[1] += xv * a.y; acc[2] += xv * a.z; acc[3] += xv * a.w;
    acc[4] += xv * c.x; acc[5] += xv * c.y; acc[6] += xv * c.z; acc[7] += xv * c.w;
  }
  for (int off = 16; off; off >>= 1)
#pragma unroll
    for (int e = 0; e < 8; ++e) acc[e] += __shfl_xor(acc[e], off, 32);
  float m = -1e30f;
#pragma unroll
  for (int e = 0; e < 8; ++e) { acc[e] += bg[e]; m = fmaxf(m, acc[e]); }
  float s = 0.f;
#pragma unroll
  for (int e = 0; e < 8; ++e) { acc[e] = __expf(acc[e] - m); s += acc[e]; }
  float inv = 1.f / s;
#pragma unroll
  for (int e = 0; e < 8; ++e) acc[e] = acc[e] * inv + ebias[e];
  float v0 = -1e30f, v1 = -1e30f; int e0 = 0, e1 = 0;
#pragma unroll
  for (int e = 0; e < 8; ++e) {
    float g = acc[e];
    if (g > v0)      { v1 = v0; e1 = e0; v0 = g; e0 = e; }
    else if (g > v1) { v1 = g; e1 = e; }
  }
  if (lane == 0) {
    gw[b * 2] = v0; gw[b * 2 + 1] = v1;
    ge[b * 2] = e0; ge[b * 2 + 1] = e1;
    atomicAdd(&cnt[e0], 1);
    atomicAdd(&cnt[8 + e1], 1);
  }
}

__global__ void zero16(int* c) { if (threadIdx.x < 16) c[threadIdx.x] = 0; }

__global__ void scan16(const int* __restrict__ cnt, int* __restrict__ off,
                       int* __restrict__ cur) {
  if (threadIdx.x == 0) {
    int s = 0;
    for (int i = 0; i < 16; ++i) { off[i] = s; cur[i] = s; s += cnt[i]; }
  }
}

__global__ void scatter_tok(const int* __restrict__ ge, int* cur,
                            int* __restrict__ list) {
  int b = blockIdx.x * 256 + threadIdx.x;
  if (b < B_TOK) {
    int e0 = ge[b * 2];     int p0 = atomicAdd(&cur[e0], 1);     list[p0] = b;
    int e1 = ge[b * 2 + 1]; int p1 = atomicAdd(&cur[8 + e1], 1); list[p1] = b;
  }
}

// ------------- grouped expert GEMM: gathered rows, bf16 WMMA 16x16x32 -------------
// 128x128 tile, 8 waves each computing 32x64. Async double-buffered LDS staging.
__global__ __launch_bounds__(256) void qkv_gemm(
    const __hip_bfloat16* __restrict__ src,  // B x 1024
    const __hip_bfloat16* __restrict__ WT,   // E x 3072 x 1024  ([e][n][k])
    const int* __restrict__ counts, const int* __restrict__ offs,
    const int* __restrict__ list, const float* __restrict__ gw,
    __hip_bfloat16* __restrict__ dst,        // B x 3072
    int pass, int nct_base, int nct_num, int accum) {
  int rem = blockIdx.x;
  int e, cnt = 0, rt = 0, ct = 0; bool found = false;
  for (e = 0; e < NEXP; ++e) {
    cnt = counts[pass * 8 + e];
    int mt = (cnt + TILE_M - 1) >> 7;
    int t = mt * nct_num;
    if (rem < t) { rt = rem / nct_num; ct = nct_base + rem % nct_num; found = true; break; }
    rem -= t;
  }
  if (!found) return;

  __shared__ alignas(16) __hip_bfloat16 As[2 * LDS_TILE];
  __shared__ alignas(16) __hip_bfloat16 Bs[2 * LDS_TILE];
  __shared__ int   tok_s[TILE_M];
  __shared__ float w_s[TILE_M];

  int tid = threadIdx.x;
  if (tid < TILE_M) {
    int gr = rt * TILE_M + tid;
    int idx = gr < cnt ? gr : cnt - 1;
    int b = list[offs[pass * 8 + e] + idx];
    tok_s[tid] = b;
    w_s[tid] = gw[b * 2 + pass];
  }
  __syncthreads();

  const __hip_bfloat16* wbase = WT + ((size_t)e * D3 + (size_t)ct * TILE_N) * DIM;

  auto stage = [&](int k0, int sel) {
    __hip_bfloat16* Asb = As + sel * LDS_TILE;
    __hip_bfloat16* Bsb = Bs + sel * LDS_TILE;
#pragma unroll
    for (int i = 0; i < 4; ++i) {
      int v = tid + 256 * i;
      int row = v >> 3, kc = (v & 7) * 8;
      async_ld16(src + (size_t)tok_s[row] * DIM + k0 + kc, Asb + row * SA + kc);
      async_ld16(wbase + (size_t)row * DIM + k0 + kc,      Bsb + row * SA + kc);
    }
  };

  int wid = tid >> 5, lane = tid & 31;
  int wm = (wid & 3) * 32, wn = (wid >> 2) * 64;
  int lm = lane & 15, hi = lane >> 4;

  v8f acc[8] = {};

  stage(0, 0);
  for (int it = 0; it < DIM / KC; ++it) {
    if (it + 1 < DIM / KC) { stage((it + 1) * KC, (it + 1) & 1); wait_async<8>(); }
    else                   { wait_async<0>(); }
    __syncthreads();
    const __hip_bfloat16* Asb = As + (it & 1) * LDS_TILE;
    const __hip_bfloat16* Bsb = Bs + (it & 1) * LDS_TILE;
#pragma unroll
    for (int ks = 0; ks < KC; ks += 32) {
      Frag a0 = ldA(Asb, (wm + lm) * SA + ks + 16 * hi);
      Frag a1 = ldA(Asb, (wm + 16 + lm) * SA + ks + 16 * hi);
      Frag b0 = ldB(Bsb, (wn + lm) * SA + ks + 16 * hi);
      Frag b1 = ldB(Bsb, (wn + 16 + lm) * SA + ks + 16 * hi);
      Frag b2 = ldB(Bsb, (wn + 32 + lm) * SA + ks + 16 * hi);
      Frag b3 = ldB(Bsb, (wn + 48 + lm) * SA + ks + 16 * hi);
      acc[0] = WMMA_BF16(a0, b0, acc[0]);
      acc[1] = WMMA_BF16(a0, b1, acc[1]);
      acc[2] = WMMA_BF16(a0, b2, acc[2]);
      acc[3] = WMMA_BF16(a0, b3, acc[3]);
      acc[4] = WMMA_BF16(a1, b0, acc[4]);
      acc[5] = WMMA_BF16(a1, b1, acc[5]);
      acc[6] = WMMA_BF16(a1, b2, acc[6]);
      acc[7] = WMMA_BF16(a1, b3, acc[7]);
    }
    __syncthreads();
  }

#pragma unroll
  for (int ar = 0; ar < 2; ++ar)
#pragma unroll
    for (int cb = 0; cb < 4; ++cb)
#pragma unroll
      for (int r = 0; r < 8; ++r) {
        int m = wm + 16 * ar + r + 8 * hi;      // C/D layout: lane row = r + 8*(lane>>4)
        if (rt * TILE_M + m < cnt) {
          int b = tok_s[m];
          float g = w_s[m];
          int col = ct * TILE_N + wn + 16 * cb + lm;
          size_t di = (size_t)b * D3 + col;
          float v = g * acc[ar * 4 + cb][r];
          if (accum) v += bf2f(dst[di]);
          dst[di] = f2bf(v);
        }
      }
}

// ---------------- per-token 8x8 head attention ----------------
__global__ __launch_bounds__(256) void attn_kernel(
    const __hip_bfloat16* __restrict__ qkvx,
    const __hip_bfloat16* __restrict__ qkvy,
    __hip_bfloat16* __restrict__ aout) {
  int wid = threadIdx.x >> 5, lane = threadIdx.x & 31;
  int b = blockIdx.x * 8 + wid;
  const __hip_bfloat16* xr = qkvx + (size_t)b * D3;
  const __hip_bfloat16* yr = qkvy + (size_t)b * D3;
  float kf[8][4], vf[8][4];
#pragma unroll
  for (int j = 0; j < 8; ++j)
#pragma unroll
    for (int c = 0; c < 4; ++c) {
      int d = lane + 32 * c;
      kf[j][c] = bf2f(xr[1024 + j * 128 + d]);
      vf[j][c] = bf2f(xr[2048 + j * 128 + d]);
    }
  const float scale = 0.088388347648318447f;  // 128^-0.5
#pragma unroll
  for (int i = 0; i < 8; ++i) {
    float q[4];
#pragma unroll
    for (int c = 0; c < 4; ++c) q[c] = bf2f(yr[i * 128 + lane + 32 * c]);
    float s[8];
#pragma unroll
    for (int j = 0; j < 8; ++j) {
      float t = q[0] * kf[j][0] + q[1] * kf[j][1] + q[2] * kf[j][2] + q[3] * kf[j][3];
      for (int off = 16; off; off >>= 1) t += __shfl_xor(t, off, 32);
      s[j] = t * scale;
    }
    float m = s[0];
#pragma unroll
    for (int j = 1; j < 8; ++j) m = fmaxf(m, s[j]);
    float sum = 0.f;
#pragma unroll
    for (int j = 0; j < 8; ++j) { s[j] = __expf(s[j] - m); sum += s[j]; }
    float inv = 1.f / sum;
#pragma unroll
    for (int c = 0; c < 4; ++c) {
      float o = 0.f;
#pragma unroll
      for (int j = 0; j < 8; ++j) o += s[j] * vf[j][c];
      // out[b, i, d] -> swapaxes -> col = d*8 + i
      aout[(size_t)b * DIM + (lane + 32 * c) * 8 + i] = f2bf(o * inv);
    }
  }
}

// ---------------- final projection GEMM + bias, f32 output ----------------
__global__ __launch_bounds__(256) void out_gemm(
    const __hip_bfloat16* __restrict__ A,   // 4096 x 1024
    const __hip_bfloat16* __restrict__ WT,  // 1024 x 1024 ([n][k])
    const float* __restrict__ bias,
    float* __restrict__ out) {
  int rt = blockIdx.x >> 3, ct = blockIdx.x & 7;
  __shared__ alignas(16) __hip_bfloat16 As[2 * LDS_TILE];
  __shared__ alignas(16) __hip_bfloat16 Bs[2 * LDS_TILE];
  int tid = threadIdx.x;
  int wid = tid >> 5, lane = tid & 31;
  int wm = (wid & 3) * 32, wn = (wid >> 2) * 64;
  int lm = lane & 15, hi = lane >> 4;

  const __hip_bfloat16* wbase = WT + (size_t)ct * TILE_N * DIM;
  const __hip_bfloat16* abase = A + (size_t)rt * TILE_M * DIM;

  auto stage = [&](int k0, int sel) {
    __hip_bfloat16* Asb = As + sel * LDS_TILE;
    __hip_bfloat16* Bsb = Bs + sel * LDS_TILE;
#pragma unroll
    for (int i = 0; i < 4; ++i) {
      int v = tid + 256 * i;
      int row = v >> 3, kc = (v & 7) * 8;
      async_ld16(abase + (size_t)row * DIM + k0 + kc, Asb + row * SA + kc);
      async_ld16(wbase + (size_t)row * DIM + k0 + kc, Bsb + row * SA + kc);
    }
  };

  v8f acc[8] = {};

  stage(0, 0);
  for (int it = 0; it < DIM / KC; ++it) {
    if (it + 1 < DIM / KC) { stage((it + 1) * KC, (it + 1) & 1); wait_async<8>(); }
    else                   { wait_async<0>(); }
    __syncthreads();
    const __hip_bfloat16* Asb = As + (it & 1) * LDS_TILE;
    const __hip_bfloat16* Bsb = Bs + (it & 1) * LDS_TILE;
#pragma unroll
    for (int ks = 0; ks < KC; ks += 32) {
      Frag a0 = ldA(Asb, (wm + lm) * SA + ks + 16 * hi);
      Frag a1 = ldA(Asb, (wm + 16 + lm) * SA + ks + 16 * hi);
      Frag b0 = ldB(Bsb, (wn + lm) * SA + ks + 16 * hi);
      Frag b1 = ldB(Bsb, (wn + 16 + lm) * SA + ks + 16 * hi);
      Frag b2 = ldB(Bsb, (wn + 32 + lm) * SA + ks + 16 * hi);
      Frag b3 = ldB(Bsb, (wn + 48 + lm) * SA + ks + 16 * hi);
      acc[0] = WMMA_BF16(a0, b0, acc[0]);
      acc[1] = WMMA_BF16(a0, b1, acc[1]);
      acc[2] = WMMA_BF16(a0, b2, acc[2]);
      acc[3] = WMMA_BF16(a0, b3, acc[3]);
      acc[4] = WMMA_BF16(a1, b0, acc[4]);
      acc[5] = WMMA_BF16(a1, b1, acc[5]);
      acc[6] = WMMA_BF16(a1, b2, acc[6]);
      acc[7] = WMMA_BF16(a1, b3, acc[7]);
    }
    __syncthreads();
  }

#pragma unroll
  for (int ar = 0; ar < 2; ++ar)
#pragma unroll
    for (int cb = 0; cb < 4; ++cb)
#pragma unroll
      for (int r = 0; r < 8; ++r) {
        int row = rt * TILE_M + wm + 16 * ar + r + 8 * hi;
        int col = ct * TILE_N + wn + 16 * cb + lm;
        out[(size_t)row * DIM + col] = acc[ar * 4 + cb][r] + bias[col];
      }
}

extern "C" void kernel_launch(void* const* d_in, const int* in_sizes, int n_in,
                              void* d_out, int out_size, void* d_ws, size_t ws_size,
                              hipStream_t stream) {
  const float* x     = (const float*)d_in[0];
  const float* y     = (const float*)d_in[1];
  const float* Wqkv  = (const float*)d_in[2];
  const float* Wg    = (const float*)d_in[3];
  const float* bg    = (const float*)d_in[4];
  const float* Wp    = (const float*)d_in[5];
  const float* bp    = (const float*)d_in[6];
  const float* ebias = (const float*)d_in[7];
  float* out = (float*)d_out;
  (void)in_sizes; (void)n_in; (void)out_size; (void)ws_size;

  char* ws = (char*)d_ws;
  size_t o = 0;
  auto take = [&](size_t bytes) {
    char* p = ws + o;
    o = (o + bytes + 255) & ~(size_t)255;
    return p;
  };
  __hip_bfloat16* x16    = (__hip_bfloat16*)take((size_t)B_TOK * DIM * 2);
  __hip_bfloat16* y16    = (__hip_bfloat16*)take((size_t)B_TOK * DIM * 2);
  __hip_bfloat16* wqkvT  = (__hip_bfloat16*)take((size_t)NEXP * D3 * DIM * 2);
  __hip_bfloat16* wpT    = (__hip_bfloat16*)take((size_t)DIM * DIM * 2);
  __hip_bfloat16* qkvx   = (__hip_bfloat16*)take((size_t)B_TOK * D3 * 2);
  __hip_bfloat16* qkvy   = (__hip_bfloat16*)take((size_t)B_TOK * D3 * 2);
  __hip_bfloat16* attn16 = (__hip_bfloat16*)take((size_t)B_TOK * DIM * 2);
  float* gw   = (float*)take((size_t)B_TOK * 2 * 4);
  int*   ge   = (int*)take((size_t)B_TOK * 2 * 4);
  int*   cnt  = (int*)take(64);
  int*   off  = (int*)take(64);
  int*   cur  = (int*)take(64);
  int*   list = (int*)take((size_t)2 * B_TOK * 4);

  // one-time (per launch) precision/layout conversions
  cvt_f32_bf16<<<4096, 256, 0, stream>>>(x, x16, B_TOK * DIM);
  cvt_f32_bf16<<<4096, 256, 0, stream>>>(y, y16, B_TOK * DIM);
  tcvt_f32_bf16<<<dim3(96, 32, 8), dim3(32, 8), 0, stream>>>(Wqkv, wqkvT, DIM, D3);
  tcvt_f32_bf16<<<dim3(32, 32, 1), dim3(32, 8), 0, stream>>>(Wp, wpT, DIM, DIM);

  // gating + expert bucketing
  zero16<<<1, 32, 0, stream>>>(cnt);
  gate_topk<<<512, 256, 0, stream>>>(x, Wg, bg, ebias, gw, ge, cnt);
  scan16<<<1, 32, 0, stream>>>(cnt, off, cur);
  scatter_tok<<<16, 256, 0, stream>>>(ge, cur, list);

  // grouped expert GEMMs (x-stream: only k,v cols [1024,3072); y-stream: only q cols [0,1024))
  qkv_gemm<<<39 * 16, 256, 0, stream>>>(x16, wqkvT, cnt, off, list, gw, qkvx, 0, 8, 16, 0);
  qkv_gemm<<<39 * 16, 256, 0, stream>>>(x16, wqkvT, cnt, off, list, gw, qkvx, 1, 8, 16, 1);
  qkv_gemm<<<39 * 8,  256, 0, stream>>>(y16, wqkvT, cnt, off, list, gw, qkvy, 0, 0, 8, 0);
  qkv_gemm<<<39 * 8,  256, 0, stream>>>(y16, wqkvT, cnt, off, list, gw, qkvy, 1, 0, 8, 1);

  // per-token cross-head attention (writes transposed layout, bf16)
  attn_kernel<<<512, 256, 0, stream>>>(qkvx, qkvy, attn16);

  // output projection
  out_gemm<<<256, 256, 0, stream>>>(attn16, wpT, bp, out);
}
